// GC_33526514712831
// MI455X (gfx1250) — matrix operand order repaired and checked by reading, not compiled, not run on hardware
//
#include <hip/hip_runtime.h>
#include <math.h>

typedef __attribute__((ext_vector_type(16))) _Float16 v16h;
typedef __attribute__((ext_vector_type(8)))  _Float16 v8h;
typedef __attribute__((ext_vector_type(8)))  float    v8f;
typedef __attribute__((ext_vector_type(4)))  int      v4i;

#define BB 8
#define CCH 64
#define NNP 4096
#define LOG2E 1.44269504088896f

#if __has_builtin(__builtin_amdgcn_global_load_async_to_lds_b128) && \
    __has_builtin(__builtin_amdgcn_s_wait_asynccnt)
#define HAVE_ASYNC_LDS 1
#else
#define HAVE_ASYNC_LDS 0
#endif

#define AS_GLOBAL __attribute__((address_space(1)))
#define AS_LDS    __attribute__((address_space(3)))

__device__ __forceinline__ float fast_exp2(float x) {
#if __has_builtin(__builtin_amdgcn_exp2f)
  return __builtin_amdgcn_exp2f(x);     // v_exp_f32 (base-2 on AMD)
#else
  return exp2f(x);
#endif
}

#if __has_builtin(__builtin_amdgcn_update_dpp)
template <int CTRL>
__device__ __forceinline__ float dpp_mov(float v) {
  return __int_as_float(
      __builtin_amdgcn_update_dpp(0, __float_as_int(v), CTRL, 0xf, 0xf, true));
}
// max over each 16-lane half-wave (VALU DPP, no LDS traffic)
__device__ __forceinline__ float rowmax16(float v) {
  v = fmaxf(v, dpp_mov<0xB1>(v));   // quad_perm [1,0,3,2]  (xor 1)
  v = fmaxf(v, dpp_mov<0x4E>(v));   // quad_perm [2,3,0,1]  (xor 2)
  v = fmaxf(v, dpp_mov<0x141>(v));  // row_half_mirror      (4s -> 8s)
  v = fmaxf(v, dpp_mov<0x140>(v));  // row_mirror           (8s -> 16)
  return v;
}
#else
__device__ __forceinline__ float rowmax16(float v) {
#pragma unroll
  for (int off = 1; off < 16; off <<= 1) v = fmaxf(v, __shfl_xor(v, off, 32));
  return v;
}
#endif

// ---------------------------------------------------------------------------
// Kernel 1: q[b][n][c] = (f16) x[b][c][n];  vt[b][c][n] = (f16) (x^T W)[n][c]
// ---------------------------------------------------------------------------
__global__ __launch_bounds__(256) void nlb_prep(const float* __restrict__ x,
                                                const float* __restrict__ w,
                                                _Float16* __restrict__ q,
                                                _Float16* __restrict__ vt) {
  __shared__ float xs[64][65];   // [c][n_local], padded
  __shared__ float Ws[64][65];   // [k][c], padded
  const int blk = blockIdx.x;
  const int b   = blk >> 6;
  const int n0  = (blk & 63) << 6;
  const int t   = threadIdx.x;

  const float* xb = x + ((size_t)b * CCH) * NNP + n0;
  for (int idx = t; idx < 64 * 64; idx += 256) {
    const int r = idx >> 6, k = idx & 63;
    xs[r][k] = xb[(size_t)r * NNP + k];
    Ws[r][k] = w[idx];
  }
  __syncthreads();

  _Float16* qb = q + ((size_t)b * NNP + n0) * CCH;
  for (int idx = t; idx < 64 * 64; idx += 256) {
    const int nn = idx >> 6, c = idx & 63;
    qb[nn * CCH + c] = (_Float16)xs[c][nn];
  }

  _Float16* vb = vt + (size_t)b * CCH * NNP + n0;
  for (int idx = t; idx < 64 * 64; idx += 256) {
    const int c = idx >> 6, nn = idx & 63;
    float acc = 0.f;
#pragma unroll
    for (int k = 0; k < 64; ++k) acc = fmaf(xs[k][nn], Ws[k][c], acc);
    vb[(size_t)c * NNP + nn] = (_Float16)acc;
  }
}

// ---------------------------------------------------------------------------
// Kernel 2: fused flash attention, 4 waves/block, 16 Q rows per wave.
// Double-buffered K/V tiles (async-to-LDS when available). Softmax in base-2.
// Row sums tracked by an all-ones WMMA column (no shuffle reduction).
// ---------------------------------------------------------------------------
__global__ __launch_bounds__(128) void nlb_attn(const float* __restrict__ x,
                                                const _Float16* __restrict__ q,
                                                const _Float16* __restrict__ vt,
                                                float* __restrict__ out) {
  __shared__ _Float16 Ks[2][64 * 72];   // K tile [j][c], pitch 72
  __shared__ _Float16 Vs[2][64 * 72];   // V tile [c][j], pitch 72 (transposed)
  __shared__ _Float16 Pl[4 * 16 * 72];  // per-wave P bounce [row][j]

  const int blk  = blockIdx.x;
  const int b    = blk >> 6;
  const int r0   = (blk & 63) << 6;
  const int tid  = threadIdx.x;
  const int lane = tid & 31;
  const int wave = tid >> 5;
  const int l16  = lane & 15;
  const int hf   = lane >> 4;
  const int i0   = r0 + wave * 16;

  // ---- Q A-fragments, pre-scaled by log2(e) so scores are base-2 ----
  const _Float16* qrow = q + ((size_t)b * NNP + (i0 + l16)) * CCH;
  v16h aq[2];
#pragma unroll
  for (int kc = 0; kc < 2; ++kc) {
    const int c1 = kc * 32 + hf * 8;
    v8h lo = *(const v8h*)(qrow + c1);
    v8h hi = *(const v8h*)(qrow + c1 + 16);
#pragma unroll
    for (int e = 0; e < 8; ++e) {
      aq[kc][e]     = (_Float16)((float)lo[e] * LOG2E);
      aq[kc][8 + e] = (_Float16)((float)hi[e] * LOG2E);
    }
  }

  v16h vones;
#pragma unroll
  for (int e = 0; e < 16; ++e) vones[e] = (_Float16)1.0f;

  v8f O[4] = {};
  v8f O4   = {};                 // row-sum accumulator (ones-column trick)
  float m[8];
#pragma unroll
  for (int r = 0; r < 8; ++r) m[r] = -INFINITY;

  const _Float16* kb_base = q  + (size_t)b * NNP * CCH;
  const _Float16* vb_base = vt + (size_t)b * CCH * NNP;
  _Float16* Pw = Pl + wave * (16 * 72);

  auto issue_tile = [&](int bufi, int j0) {
    const _Float16* kb = kb_base + (size_t)j0 * CCH;
    const _Float16* vb = vb_base + j0;
#pragma unroll
    for (int u = 0; u < 4; ++u) {
      const int q8  = tid + u * 128;
      const int row = q8 >> 3, seg = (q8 & 7) << 3;
#if HAVE_ASYNC_LDS
      __builtin_amdgcn_global_load_async_to_lds_b128(
          (AS_GLOBAL v4i*)(kb + (size_t)row * CCH + seg),
          (AS_LDS v4i*)&Ks[bufi][row * 72 + seg], 0, 0);
      __builtin_amdgcn_global_load_async_to_lds_b128(
          (AS_GLOBAL v4i*)(vb + (size_t)row * NNP + seg),
          (AS_LDS v4i*)&Vs[bufi][row * 72 + seg], 0, 0);
#else
      *(v8h*)&Ks[bufi][row * 72 + seg] = *(const v8h*)(kb + (size_t)row * CCH + seg);
      *(v8h*)&Vs[bufi][row * 72 + seg] = *(const v8h*)(vb + (size_t)row * NNP + seg);
#endif
    }
  };

  int buf = 0;
  issue_tile(0, 0);

  for (int j0 = 0; j0 < NNP; j0 += 64) {
    if (j0 + 64 < NNP) {
      issue_tile(buf ^ 1, j0 + 64);      // prefetch next tile into other buffer
#if HAVE_ASYNC_LDS
      __builtin_amdgcn_s_wait_asynccnt(8);  // current tile (8 older asyncs) done
#endif
    } else {
#if HAVE_ASYNC_LDS
      __builtin_amdgcn_s_wait_asynccnt(0);
#endif
    }
    __syncthreads();

    const _Float16* KsB = Ks[buf];
    const _Float16* VsB = Vs[buf];

    // ---- S = Q_strip (16x64) * K_tile^T, in base-2 score domain ----
    v8f S[4];
#pragma unroll
    for (int jj = 0; jj < 4; ++jj) {
      v8f s = {};
#pragma unroll
      for (int kc = 0; kc < 2; ++kc) {
        const _Float16* bp = &KsB[(jj * 16 + l16) * 72 + kc * 32 + hf * 8];
        v8h b0 = *(const v8h*)bp;
        v8h b1 = *(const v8h*)(bp + 16);
        v16h bk;
#pragma unroll
        for (int e = 0; e < 8; ++e) { bk[e] = b0[e]; bk[8 + e] = b1[e]; }
        s = __builtin_amdgcn_wmma_f32_16x16x32_f16(false, aq[kc], false, bk,
                                                   (short)0, s, false, false);
      }
      S[jj] = s;
    }

    // ---- online softmax: block max via DPP, P = exp2(S - m) ----
    float sc[8];
#pragma unroll
    for (int r = 0; r < 8; ++r) {
      float mb = fmaxf(fmaxf(S[0][r], S[1][r]), fmaxf(S[2][r], S[3][r]));
      mb = rowmax16(mb);
      const float mn = fmaxf(m[r], mb);
      sc[r] = fast_exp2(m[r] - mn);
      m[r]  = mn;
    }
#pragma unroll
    for (int jj = 0; jj < 4; ++jj)
#pragma unroll
      for (int r = 0; r < 8; ++r)
        S[jj][r] = fast_exp2(S[jj][r] - m[r]);

#pragma unroll
    for (int cc = 0; cc < 4; ++cc)
#pragma unroll
      for (int r = 0; r < 8; ++r) O[cc][r] *= sc[r];
#pragma unroll
    for (int r = 0; r < 8; ++r) O4[r] *= sc[r];

    // ---- repack P (C-layout regs -> A-layout) via per-wave LDS bounce ----
#pragma unroll
    for (int jj = 0; jj < 4; ++jj)
#pragma unroll
      for (int r = 0; r < 8; ++r)
        Pw[(r + hf * 8) * 72 + jj * 16 + l16] = (_Float16)S[jj][r];

    // ---- O += P * V ; O4 += P * ones (softmax denominator) ----
#pragma unroll
    for (int kk = 0; kk < 2; ++kk) {
      const _Float16* pp = Pw + l16 * 72 + kk * 32 + hf * 8;
      v8h p0 = *(const v8h*)pp;
      v8h p1 = *(const v8h*)(pp + 16);
      v16h pa;
#pragma unroll
      for (int e = 0; e < 8; ++e) { pa[e] = p0[e]; pa[8 + e] = p1[e]; }
#pragma unroll
      for (int cc = 0; cc < 4; ++cc) {
        const _Float16* bp = &VsB[(cc * 16 + l16) * 72 + kk * 32 + hf * 8];
        v8h b0 = *(const v8h*)bp;
        v8h b1 = *(const v8h*)(bp + 16);
        v16h bv;
#pragma unroll
        for (int e = 0; e < 8; ++e) { bv[e] = b0[e]; bv[8 + e] = b1[e]; }
        O[cc] = __builtin_amdgcn_wmma_f32_16x16x32_f16(false, pa, false, bv,
                                                       (short)0, O[cc], false, false);
      }
      O4 = __builtin_amdgcn_wmma_f32_16x16x32_f16(false, pa, false, vones,
                                                  (short)0, O4, false, false);
    }

    __syncthreads();   // all waves done reading buf before it is refilled
    buf ^= 1;
  }

  // ---- epilogue: normalize, transpose back via D layout, add residual ----
  float inv[8];
#pragma unroll
  for (int r = 0; r < 8; ++r) inv[r] = 1.0f / O4[r];

  const size_t ob = (size_t)b * CCH * NNP;
#pragma unroll
  for (int cc = 0; cc < 4; ++cc) {
    const int c    = cc * 16 + l16;
    const int irow = i0 + hf * 8;
    float*       op = out + ob + (size_t)c * NNP + irow;
    const float* xp = x   + ob + (size_t)c * NNP + irow;
#pragma unroll
    for (int r = 0; r < 8; ++r)
      op[r] = O[cc][r] * inv[r] + xp[r];
  }
}

// ---------------------------------------------------------------------------
extern "C" void kernel_launch(void* const* d_in, const int* in_sizes, int n_in,
                              void* d_out, int out_size, void* d_ws, size_t ws_size,
                              hipStream_t stream) {
  (void)in_sizes; (void)n_in; (void)out_size; (void)ws_size;
  const float* x = (const float*)d_in[0];
  const float* w = (const float*)d_in[1];
  float* out = (float*)d_out;

  _Float16* q  = (_Float16*)d_ws;                      // 4 MB: [B][N][C] f16
  _Float16* vt = q + (size_t)BB * NNP * CCH;           // 4 MB: [B][C][N] f16

  nlb_prep<<<BB * (NNP / 64), 256, 0, stream>>>(x, w, q, vt);
  nlb_attn<<<BB * (NNP / 64), 128, 0, stream>>>(x, q, vt, out);
}
